// FeatureEmbedded_65266323030193
// MI455X (gfx1250) — compile-verified, matrix operand order checked by reading, not run on hardware
//
#include <hip/hip_runtime.h>
#include <hip/hip_bf16.h>

#define I_DIM 192
#define H_DIM 512
#define T_DIM 1024
#define B_DIM 64
#define G3H   1536            // 3*H
#define MROWS (T_DIM * B_DIM) // 65536
#define HPAD  520             // LDS row stride (bf16 elems) to dodge bank conflicts

typedef __attribute__((ext_vector_type(16))) __bf16 v16bf;
typedef __attribute__((ext_vector_type(8)))  float  v8f;
typedef __attribute__((ext_vector_type(4)))  unsigned v4u;
typedef __attribute__((ext_vector_type(8)))  int    v8i;
typedef __attribute__((ext_vector_type(4)))  int    v4i;

union ABPack { v16bf v; unsigned u[8]; };

__device__ __forceinline__ unsigned ld_pair(const __hip_bfloat16* p) {
  return *(const unsigned*)p; // two consecutive bf16 (even k, even row stride -> 4B aligned)
}

// 16x32 bf16 tile (A or B operand) from a row-major [rows, ldk] matrix.
// Documented CDNA5 16-bit operand layout: lane lo=lane&15 picks row (M or N),
// lane hi=lane>>4 adds K offset 8; VGPR v in 0..3 -> K = 2v, v in 4..7 -> K = 16+2(v-4).
__device__ __forceinline__ v16bf load_tile16x32(const __hip_bfloat16* __restrict__ base,
                                                int row0, int k0, int ldk) {
  const int lane = threadIdx.x & 31;
  const int lo = lane & 15, hi = lane >> 4;
  const __hip_bfloat16* r = base + (size_t)(row0 + lo) * ldk + (k0 + hi * 8);
  ABPack p;
  p.u[0] = ld_pair(r + 0);
  p.u[1] = ld_pair(r + 2);
  p.u[2] = ld_pair(r + 4);
  p.u[3] = ld_pair(r + 6);
  p.u[4] = ld_pair(r + 16);
  p.u[5] = ld_pair(r + 18);
  p.u[6] = ld_pair(r + 20);
  p.u[7] = ld_pair(r + 22);
  return p.v;
}

// Same tile fetch but from LDS h[64][HPAD]
__device__ __forceinline__ v16bf load_tile_lds(const __hip_bfloat16* hbase, int m0, int k0) {
  const int lane = threadIdx.x & 31;
  const int lo = lane & 15, hi = lane >> 4;
  const __hip_bfloat16* r = hbase + (m0 + lo) * HPAD + (k0 + hi * 8);
  ABPack p;
  p.u[0] = ld_pair(r + 0);
  p.u[1] = ld_pair(r + 2);
  p.u[2] = ld_pair(r + 4);
  p.u[3] = ld_pair(r + 6);
  p.u[4] = ld_pair(r + 16);
  p.u[5] = ld_pair(r + 18);
  p.u[6] = ld_pair(r + 20);
  p.u[7] = ld_pair(r + 22);
  return p.v;
}

__device__ __forceinline__ v8f wmma_bf16(v16bf a, v16bf b, v8f c) {
  return __builtin_amdgcn_wmma_f32_16x16x32_bf16(false, a, false, b, (short)0, c, false, false);
}

__device__ __forceinline__ float sigmoidf_(float x) {
  return 1.0f / (1.0f + __expf(-x));
}

// Issue a TDM 2D tile load: [rows x cols] bf16, row-major contiguous (stride = cols),
// global -> LDS. Wave-uniform arguments; one issuing wave, tracked by TENSORcnt.
__device__ __forceinline__ void tdm_load_tile_bf16(const __hip_bfloat16* gsrc,
                                                   unsigned lds_off, int cols, int rows) {
  unsigned long long ga = (unsigned long long)(uintptr_t)gsrc;
  v4u g0;
  g0.x = 1u;                                  // count=1 (valid), is_restore=0, gather=0
  g0.y = lds_off;                             // lds_addr (bytes)
  g0.z = (unsigned)ga;                        // global_addr[31:0]
  g0.w = (unsigned)((ga >> 32) & 0x01FFFFFFu) // global_addr[56:32]
         | 0x80000000u;                       // type=2 ("image") in bits[127:126]
  v8i g1;
  g1[0] = (1 << 16);                          // workgroup_mask=0, data_size=1 (2 bytes)
  g1[1] = (cols & 0xFFFF) << 16;              // tensor_dim0[15:0] at bits[63:48]
  g1[2] = ((cols >> 16) & 0xFFFF) | ((rows & 0xFFFF) << 16); // dim0 hi, tensor_dim1 lo
  g1[3] = (cols & 0xFFFF) << 16;              // tensor_dim1 hi = 0, tile_dim0 at bits[127:112]
  g1[4] = rows & 0xFFFF;                      // tile_dim1; tile_dim2 = 0
  g1[5] = cols;                               // tensor_dim0_stride[31:0]
  g1[6] = 0;                                  // stride hi, tensor_dim1_stride lo
  g1[7] = 0;                                  // tensor_dim1_stride hi
  v4i z4 = {0, 0, 0, 0};
#if defined(__clang_major__) && (__clang_major__ >= 23)
  v8i z8 = {0, 0, 0, 0, 0, 0, 0, 0};
  __builtin_amdgcn_tensor_load_to_lds(g0, g1, z4, z4, z8, 0);
#else
  __builtin_amdgcn_tensor_load_to_lds(g0, g1, z4, z4, 0);
#endif
}

// ---------------- conversion kernels ----------------

// x [B,T,I] f32 -> xa [(t*B+b), I] bf16 (t-major rows, GEMM-ready)
__global__ void convert_x_kernel(const float* __restrict__ x, __hip_bfloat16* __restrict__ xa) {
  int idx = blockIdx.x * blockDim.x + threadIdx.x; // exactly MROWS*I_DIM threads
  int i = idx % I_DIM;
  int r = idx / I_DIM;       // r = t*B + b
  int b = r & (B_DIM - 1);
  int t = r >> 6;
  xa[idx] = __float2bfloat16(x[((size_t)b * T_DIM + t) * I_DIM + i]);
}

__global__ void convert_w_kernel(const float* __restrict__ w, __hip_bfloat16* __restrict__ o, int n) {
  int idx = blockIdx.x * blockDim.x + threadIdx.x;
  if (idx < n) o[idx] = __float2bfloat16(w[idx]);
}

// ---------------- input-gate GEMM:  C[M,1536] = A[M,K] * W[1536,K]^T + bias ----------------
// block = 256 threads = 8 waves laid out 4(m) x 2(n); wave computes 16x64.
__global__ __launch_bounds__(256) void gi_gemm_kernel(
    const __hip_bfloat16* __restrict__ A,
    const __hip_bfloat16* __restrict__ W,
    const float* __restrict__ bias,
    __hip_bfloat16* __restrict__ C,
    int K) {
  const int lane = threadIdx.x & 31;
  const int lo = lane & 15, hi = lane >> 4;
  const int wave = threadIdx.x >> 5;
  const int wm = wave >> 1;            // 0..3
  const int wn = wave & 1;             // 0..1
  const int row0 = blockIdx.y * 64 + wm * 16;
  const int n0 = blockIdx.x * 128 + wn * 64;

  v8f acc[4] = {};
  for (int k0 = 0; k0 < K; k0 += 32) {
    v16bf a = load_tile16x32(A, row0, k0, K);
#pragma unroll
    for (int jn = 0; jn < 4; ++jn) {
      v16bf b = load_tile16x32(W, n0 + jn * 16, k0, K);
      acc[jn] = wmma_bf16(a, b, acc[jn]);
    }
  }
#pragma unroll
  for (int jn = 0; jn < 4; ++jn) {
    const int n = n0 + jn * 16 + lo;
    const float bv = bias[n];
#pragma unroll
    for (int e = 0; e < 8; ++e) {
      const int row = row0 + e + 8 * hi; // C/D layout: VGPR e holds M=e (lanes 0-15) / e+8
      C[(size_t)row * G3H + n] = __float2bfloat16(acc[jn][e] + bv);
    }
  }
}

// ---------------- persistent recurrent scan ----------------
// grid = 2 blocks (dir 0 = fwd, 1 = bwd), 1024 threads = 32 waves = 4 m-tiles x 8 j-groups.
// Each wave owns 4 j-tiles across all 3 gates (12 WMMA accumulators) -> gate fusion is wave-local.
// Per step: wave 0 issues a TDM load of the step's gi tile into LDS (overlapped with the
// recurrent WMMA GEMM), waits on TENSORcnt, then all waves do the gate epilogue from LDS.
__global__ __launch_bounds__(1024) void gru_scan_kernel(
    const __hip_bfloat16* __restrict__ gi_f, const __hip_bfloat16* __restrict__ gi_b,
    const __hip_bfloat16* __restrict__ whh_f, const __hip_bfloat16* __restrict__ whh_b,
    const float* __restrict__ bhh_f, const float* __restrict__ bhh_b,
    const int* __restrict__ lengths,
    __hip_bfloat16* __restrict__ y_out,  // [T*B, 2H] bf16 or nullptr (layer 1)
    float* __restrict__ h_final) {       // [B, 2H] f32 or nullptr (layer 0)
  __shared__ __hip_bfloat16 h_lds[B_DIM][HPAD];            // 66.5 KB
  __shared__ __hip_bfloat16 gi_lds[B_DIM * G3H];           // 192 KB (TDM destination)

  const int dir = blockIdx.x;
  const __hip_bfloat16* gi  = dir ? gi_b : gi_f;
  const __hip_bfloat16* whh = dir ? whh_b : whh_f;
  const float* bhh          = dir ? bhh_b : bhh_f;

  const int tid = threadIdx.x;
  const int lane = tid & 31;
  const int lo = lane & 15, hi = lane >> 4;
  const int wave = tid >> 5;
  const int m_tile = wave >> 3;  // 0..3 (16 batch rows each)
  const int wj = wave & 7;       // 0..7 (4 hidden-column tiles each)
  const int m0 = m_tile * 16;
  const unsigned gi_lds_off = (unsigned)(uintptr_t)&gi_lds[0]; // LDS byte offset (low 32 bits)

  // h0 = 0
  for (int idx = tid; idx < B_DIM * HPAD; idx += 1024)
    ((__hip_bfloat16*)h_lds)[idx] = __float2bfloat16(0.0f);

  // per-lane invariants: hidden column j per j-tile, b_hh gate biases, batch rows + lengths
  int jcol[4]; float br[4], bz[4], bn[4];
#pragma unroll
  for (int jt = 0; jt < 4; ++jt) {
    int j = (wj * 4 + jt) * 16 + lo;
    jcol[jt] = j;
    br[jt] = bhh[j];
    bz[jt] = bhh[H_DIM + j];
    bn[jt] = bhh[2 * H_DIM + j];
  }
  int b_e[8], len_e[8];
#pragma unroll
  for (int e = 0; e < 8; ++e) {
    b_e[e] = m0 + e + 8 * hi;
    len_e[e] = lengths[b_e[e]];
  }
  __syncthreads();

  for (int t = 0; t < T_DIM; ++t) {
    const int trow = dir ? (T_DIM - 1 - t) : t;

    // Kick off the async gi tile DMA for this step (overlaps with the GEMM below).
    if (wave == 0)
      tdm_load_tile_bf16(gi + (size_t)trow * B_DIM * G3H, gi_lds_off, G3H, B_DIM);

    // gh = h * w_hh^T : 64x1536x512, K in 16 steps of 32
    v8f acc[4][3] = {};
    for (int k0 = 0; k0 < H_DIM; k0 += 32) {
      v16bf a = load_tile_lds(&h_lds[0][0], m0, k0);
#pragma unroll
      for (int jt = 0; jt < 4; ++jt) {
        const int n_base = (wj * 4 + jt) * 16;
#pragma unroll
        for (int g = 0; g < 3; ++g) {
          v16bf bm = load_tile16x32(whh, g * H_DIM + n_base, k0, H_DIM);
          acc[jt][g] = wmma_bf16(a, bm, acc[jt][g]);
        }
      }
    }
    if (wave == 0)
      __builtin_amdgcn_s_wait_tensorcnt(0);  // gi tile resident in LDS
    __syncthreads(); // h reads done + gi tile published to all waves

#pragma unroll
    for (int jt = 0; jt < 4; ++jt) {
      const int j = jcol[jt];
#pragma unroll
      for (int e = 0; e < 8; ++e) {
        const int b = b_e[e];
        const __hip_bfloat16* grow = &gi_lds[b * G3H];
        float ir  = __bfloat162float(grow[j]);
        float iz  = __bfloat162float(grow[H_DIM + j]);
        float inn = __bfloat162float(grow[2 * H_DIM + j]);
        float hr = acc[jt][0][e] + br[jt];
        float hz = acc[jt][1][e] + bz[jt];
        float hn = acc[jt][2][e] + bn[jt];
        float r = sigmoidf_(ir + hr);
        float z = sigmoidf_(iz + hz);
        float n = tanhf(inn + r * hn);
        float hold = __bfloat162float(h_lds[b][j]);
        float hnew = (1.0f - z) * n + z * hold;
        float h2 = (trow < len_e[e]) ? hnew : hold;   // mask freezes padded steps
        __hip_bfloat16 hb = __float2bfloat16(h2);
        h_lds[b][j] = hb;
        if (y_out)
          y_out[((size_t)trow * B_DIM + b) * (2 * H_DIM) + dir * H_DIM + j] = hb;
      }
    }
    __syncthreads(); // h writes + gi reads complete before next step's GEMM / TDM
  }

  if (h_final) { // layer-2 final hidden -> d_out [B, 2H] f32
    for (int idx = tid; idx < B_DIM * H_DIM; idx += 1024) {
      int b = idx >> 9, j = idx & (H_DIM - 1);
      h_final[(size_t)b * (2 * H_DIM) + dir * H_DIM + j] = __bfloat162float(h_lds[b][j]);
    }
  }
}

// ---------------- host ----------------

extern "C" void kernel_launch(void* const* d_in, const int* in_sizes, int n_in,
                              void* d_out, int out_size, void* d_ws, size_t ws_size,
                              hipStream_t stream) {
  (void)in_sizes; (void)n_in; (void)out_size; (void)ws_size;
  const float* x       = (const float*)d_in[0];
  const int*   lengths = (const int*)d_in[1];
  // order per dir: (w_ih, w_hh, b_ih, b_hh); dirs: l0f, l0b, l1f, l1b
  const float* w_ih[4] = {(const float*)d_in[2], (const float*)d_in[6],
                          (const float*)d_in[10], (const float*)d_in[14]};
  const float* w_hh[4] = {(const float*)d_in[3], (const float*)d_in[7],
                          (const float*)d_in[11], (const float*)d_in[15]};
  const float* b_ih[4] = {(const float*)d_in[4], (const float*)d_in[8],
                          (const float*)d_in[12], (const float*)d_in[16]};
  const float* b_hh[4] = {(const float*)d_in[5], (const float*)d_in[9],
                          (const float*)d_in[13], (const float*)d_in[17]};

  char* ws = (char*)d_ws;
  size_t off = 0;
  auto alloc = [&](size_t bytes) -> void* {
    void* p = ws + off;
    off += (bytes + 255) & ~(size_t)255;
    return p;
  };

  __hip_bfloat16* xa    = (__hip_bfloat16*)alloc((size_t)MROWS * I_DIM * 2);
  __hip_bfloat16* wih_bf[4];
  wih_bf[0] = (__hip_bfloat16*)alloc((size_t)G3H * I_DIM * 2);
  wih_bf[1] = (__hip_bfloat16*)alloc((size_t)G3H * I_DIM * 2);
  wih_bf[2] = (__hip_bfloat16*)alloc((size_t)G3H * 1024 * 2);
  wih_bf[3] = (__hip_bfloat16*)alloc((size_t)G3H * 1024 * 2);
  __hip_bfloat16* whh_bf[4];
  for (int d = 0; d < 4; ++d) whh_bf[d] = (__hip_bfloat16*)alloc((size_t)G3H * H_DIM * 2);
  __hip_bfloat16* gi_f = (__hip_bfloat16*)alloc((size_t)MROWS * G3H * 2);
  __hip_bfloat16* gi_b = (__hip_bfloat16*)alloc((size_t)MROWS * G3H * 2);
  __hip_bfloat16* y1   = (__hip_bfloat16*)alloc((size_t)MROWS * 2 * H_DIM * 2);

  // --- conversions ---
  convert_x_kernel<<<(MROWS * I_DIM) / 256, 256, 0, stream>>>(x, xa);
  const int nwih0 = G3H * I_DIM, nwih1 = G3H * 1024, nwhh = G3H * H_DIM;
  convert_w_kernel<<<(nwih0 + 255) / 256, 256, 0, stream>>>(w_ih[0], wih_bf[0], nwih0);
  convert_w_kernel<<<(nwih0 + 255) / 256, 256, 0, stream>>>(w_ih[1], wih_bf[1], nwih0);
  convert_w_kernel<<<(nwih1 + 255) / 256, 256, 0, stream>>>(w_ih[2], wih_bf[2], nwih1);
  convert_w_kernel<<<(nwih1 + 255) / 256, 256, 0, stream>>>(w_ih[3], wih_bf[3], nwih1);
  for (int d = 0; d < 4; ++d)
    convert_w_kernel<<<(nwhh + 255) / 256, 256, 0, stream>>>(w_hh[d], whh_bf[d], nwhh);

  dim3 ggrid(G3H / 128, MROWS / 64);

  // --- layer 0 ---
  gi_gemm_kernel<<<ggrid, 256, 0, stream>>>(xa, wih_bf[0], b_ih[0], gi_f, I_DIM);
  gi_gemm_kernel<<<ggrid, 256, 0, stream>>>(xa, wih_bf[1], b_ih[1], gi_b, I_DIM);
  gru_scan_kernel<<<2, 1024, 0, stream>>>(gi_f, gi_b, whh_bf[0], whh_bf[1],
                                          b_hh[0], b_hh[1], lengths, y1, nullptr);

  // --- layer 1 (reuse gi buffers) ---
  gi_gemm_kernel<<<ggrid, 256, 0, stream>>>(y1, wih_bf[2], b_ih[2], gi_f, 1024);
  gi_gemm_kernel<<<ggrid, 256, 0, stream>>>(y1, wih_bf[3], b_ih[3], gi_b, 1024);
  gru_scan_kernel<<<2, 1024, 0, stream>>>(gi_f, gi_b, whh_bf[2], whh_bf[3],
                                          b_hh[2], b_hh[3], lengths, nullptr, (float*)d_out);
}
